// RNN_66477503807913
// MI455X (gfx1250) — matrix operand order repaired and checked by reading, not compiled, not run on hardware
//
#include <hip/hip_runtime.h>

// ---------------- problem constants (from reference setup) ----------------
#define HID   256          // H
#define G3    768          // 3*H
#define TSTEPS 1024        // T
#define ROWS  16           // B-rows per block (one WMMA M-tile)
#define GH_PITCH (G3 + 4)  // pad LDS gh rows to decouple banks

typedef __attribute__((ext_vector_type(16))) __bf16 v16bf;
typedef __attribute__((ext_vector_type(8)))  __bf16 v8bf;
typedef __attribute__((ext_vector_type(8)))  float  v8f;

// ---------------- prologue: w_hh fp32 -> bf16 row-major in workspace -------
__global__ void whh_to_bf16(const float* __restrict__ w, __bf16* __restrict__ o, int n) {
    int i = blockIdx.x * blockDim.x + threadIdx.x;
    if (i < n) o[i] = (__bf16)w[i];
}

// ---------------- main persistent-recurrence kernel ------------------------
// grid = B/16 blocks, 256 threads (8 waves). Each wave owns a 96-wide slice
// of the 768 gate columns; its w_hh slice lives in 384 VGPRs for the whole
// T loop. h state (fp32 + bf16 mirror) lives in LDS for the whole T loop.
// The 6 accumulator tiles are processed as 2 passes of 3 to keep peak VGPR
// pressure (384 B + 64 A + 24 acc + temps) under the 512/wave ceiling.
__global__ __launch_bounds__(256, 1)
void rnn_gru_kernel(const float* __restrict__ x,
                    const float* __restrict__ w_ih,
                    const float* __restrict__ b_ih,
                    const float* __restrict__ b_hh,
                    const float* __restrict__ w_out,
                    const float* __restrict__ b_out,
                    const __bf16* __restrict__ whh_bf,
                    float* __restrict__ out,
                    int T)
{
    __shared__ float  lds_gh[ROWS][GH_PITCH]; // f32 gate pre-activations
    __shared__ float  lds_h [ROWS][HID];      // f32 hidden state
    __shared__ __bf16 lds_hb[ROWS][HID];      // bf16 mirror (WMMA A source)
    __shared__ float  lds_wi[G3];
    __shared__ float  lds_bi[G3];
    __shared__ float  lds_bh[G3];
    __shared__ float  lds_wo[HID];

    const int tid  = threadIdx.x;
    const int wave = tid >> 5;
    const int lane = tid & 31;
    const int b0   = blockIdx.x * ROWS;

    // stage small parameters; zero initial hidden state
    for (int i = tid; i < G3; i += 256) {
        lds_wi[i] = w_ih[i];
        lds_bi[i] = b_ih[i];
        lds_bh[i] = b_hh[i];
    }
    for (int i = tid; i < HID; i += 256) lds_wo[i] = w_out[i];
    for (int i = tid; i < ROWS * HID; i += 256) {
        (&lds_h [0][0])[i] = 0.0f;
        (&lds_hb[0][0])[i] = (__bf16)0.0f;
    }

    // ---- load this wave's w_hh slice into registers, once, for all T steps.
    // B-operand (32x16 bf16): lanes 0-15 hold K {0..7,16..23} of column N=lane,
    // lanes 16-31 hold K {8..15,24..31} of column N=lane-16. Column N of the matmul
    // h @ w_hh.T is row N of w_hh, so each lane reads two runs of 8 bf16.
    v16bf Bfrag[6][8];
    {
        const int halfK = (lane < 16) ? 0 : 8;
        #pragma unroll
        for (int nt = 0; nt < 6; ++nt) {
            const int n = (wave * 6 + nt) * 16 + (lane & 15);
            #pragma unroll
            for (int kt = 0; kt < 8; ++kt) {
                Bfrag[nt][kt] = __builtin_shufflevector(
                    *(const v8bf*)(whh_bf + n * HID + kt * 32 + halfK),
                    *(const v8bf*)(whh_bf + n * HID + kt * 32 + halfK + 16),
                    0,1,2,3,4,5,6,7,8,9,10,11,12,13,14,15);
            }
        }
    }
    __syncthreads();

    const int   arow   = lane & 15;
    const int   grow   = tid >> 4;   // gating: 16 threads per B-row
    const int   gcol   = tid & 15;
    const float bout_s = b_out[0];

    for (int t = 0; t < T; ++t) {
        // -------- GEMM phase: gh = h @ w_hh.T (bf16 WMMA, f32 accum) -------
        // A-operand (16x32 bf16): lane<16 holds row M=lane, K {0..7,16..23};
        // lane>=16 holds row M=lane-16, K {8..15,24..31}.
        v16bf A[8];
        {
            const int koff = (lane < 16) ? 0 : 8;
            #pragma unroll
            for (int kt = 0; kt < 8; ++kt) {
                const int k0 = kt * 32 + koff;
                v8bf lo = *(const v8bf*)&lds_hb[arow][k0];
                v8bf hi = *(const v8bf*)&lds_hb[arow][k0 + 16];
                A[kt] = __builtin_shufflevector(lo, hi,
                        0,1,2,3,4,5,6,7,8,9,10,11,12,13,14,15);
            }
        }

        // two passes of 3 independent accumulation chains (peak-VGPR control)
        const int mbase = (lane < 16) ? 0 : 8;
        #pragma unroll
        for (int half = 0; half < 2; ++half) {
            v8f acc[3] = {};
            #pragma unroll
            for (int kt = 0; kt < 8; ++kt) {
                #pragma unroll
                for (int q = 0; q < 3; ++q) {
                    acc[q] = __builtin_amdgcn_wmma_f32_16x16x32_bf16(
                        /*neg_a=*/false, A[kt],
                        /*neg_b=*/false, Bfrag[half * 3 + q][kt],
                        /*c_mod=*/(short)0, acc[q],
                        /*reuse_a=*/false, /*reuse_b=*/false);
                }
            }
            // scatter C tiles (f32 16x16: lane half -> M 0..7/8..15, N=lane&15)
            #pragma unroll
            for (int q = 0; q < 3; ++q) {
                const int ncol = (wave * 6 + half * 3 + q) * 16 + (lane & 15);
                #pragma unroll
                for (int v = 0; v < 8; ++v)
                    lds_gh[mbase + v][ncol] = acc[q][v];
            }
        }
        __syncthreads();

        // -------- gating + output phase (elementwise, fp32) ----------------
        {
            const float xv = x[(size_t)(b0 + grow) * T + t];
            float partial = 0.0f;
            #pragma unroll
            for (int i = 0; i < 16; ++i) {
                const int j = gcol + i * 16;
                const float gir = fmaf(xv, lds_wi[j          ], lds_bi[j          ]);
                const float giz = fmaf(xv, lds_wi[j +     HID], lds_bi[j +     HID]);
                const float gin = fmaf(xv, lds_wi[j + 2 * HID], lds_bi[j + 2 * HID]);
                const float ghr = lds_gh[grow][j          ] + lds_bh[j          ];
                const float ghz = lds_gh[grow][j +     HID] + lds_bh[j +     HID];
                const float ghn = lds_gh[grow][j + 2 * HID] + lds_bh[j + 2 * HID];
                const float r  = 1.0f / (1.0f + __expf(-(gir + ghr)));
                const float z  = 1.0f / (1.0f + __expf(-(giz + ghz)));
                const float nn = tanhf(gin + r * ghn);
                const float ho = lds_h[grow][j];
                const float hn = (1.0f - z) * nn + z * ho;
                lds_h [grow][j] = hn;
                lds_hb[grow][j] = (__bf16)hn;
                partial = fmaf(fmaxf(hn, 0.0f), lds_wo[j], partial);
            }
            // reduce the out_t dot across the 16 lanes of this row group
            #pragma unroll
            for (int s = 8; s >= 1; s >>= 1)
                partial += __shfl_xor(partial, s, 16);
            if (gcol == 0)
                out[(size_t)(b0 + grow) * T + t] = partial + bout_s;
        }
        __syncthreads();  // protect lds_hb / lds_gh for the next step
    }
}

// ---------------- harness entry -------------------------------------------
extern "C" void kernel_launch(void* const* d_in, const int* in_sizes, int n_in,
                              void* d_out, int out_size, void* d_ws, size_t ws_size,
                              hipStream_t stream)
{
    const float* x     = (const float*)d_in[0];
    const float* w_ih  = (const float*)d_in[1];
    const float* w_hh  = (const float*)d_in[2];
    const float* b_ih  = (const float*)d_in[3];
    const float* b_hh  = (const float*)d_in[4];
    const float* w_out = (const float*)d_in[5];
    const float* b_out = (const float*)d_in[6];
    float*       out   = (float*)d_out;
    __bf16*      whhbf = (__bf16*)d_ws;

    const int T = TSTEPS;
    const int B = in_sizes[0] / T;          // x is (B, T)
    const int nW = in_sizes[2];             // 3H*H

    whh_to_bf16<<<(nW + 255) / 256, 256, 0, stream>>>(w_hh, whhbf, nW);
    rnn_gru_kernel<<<B / ROWS, 256, 0, stream>>>(
        x, w_ih, b_ih, b_hh, w_out, b_out, whhbf, out, T);
}
